// AutoGraphModel_MKL_17514876633765
// MI455X (gfx1250) — matrix operand order repaired and checked by reading, not compiled, not run on hardware
//
#include <hip/hip_runtime.h>
#include <hip/hip_bf16.h>

#define BGR 8

typedef __attribute__((ext_vector_type(16))) __bf16 v16bf;
typedef __attribute__((ext_vector_type(8)))  float  v8f;
typedef __attribute__((ext_vector_type(4)))  unsigned uv4;
typedef __attribute__((ext_vector_type(8)))  int      iv8;
typedef __attribute__((ext_vector_type(4)))  int      iv4;

// native bf16 convert (RTNE on CDNA5); inputs are finite here
__device__ __forceinline__ __bf16 f2bf(float x) { return (__bf16)x; }

__device__ __forceinline__ unsigned short f2bfu(float x) {
    return __builtin_bit_cast(unsigned short, (__bf16)x);
}

__device__ __forceinline__ float elu1(float v) {
    return v > 0.f ? v : (expf(v) - 1.f);
}

// ---------------- graph-conv pipeline ----------------

__global__ void k_init(float* in_deg, float* out_deg, float* hg, float* cnt, int n) {
    int i = blockIdx.x * blockDim.x + threadIdx.x;
    if (i < n) { in_deg[i] = 1.f; out_deg[i] = 1.f; }   // self loop
    if (i < BGR * 128) hg[i] = 0.f;
    if (i < BGR) cnt[i] = 0.f;
}

__global__ void k_deg(const int* __restrict__ src, const int* __restrict__ dst,
                      float* in_deg, float* out_deg, int e) {
    int i = blockIdx.x * blockDim.x + threadIdx.x;
    if (i < e) {
        atomicAdd(&out_deg[src[i]], 1.f);
        atomicAdd(&in_deg[dst[i]], 1.f);
    }
}

__global__ void k_nodeprep(const float* __restrict__ in_deg, const float* __restrict__ out_deg,
                           const int* __restrict__ n2g,
                           float* dn_in, float* dn_out, float* x0, float* agg1,
                           float* cnt, int n) {
    __shared__ float lc[BGR];
    int t = threadIdx.x;
    if (t < BGR) lc[t] = 0.f;
    __syncthreads();
    int i = blockIdx.x * blockDim.x + t;
    if (i < n) {
        float di = in_deg[i], dg = out_deg[i];
        float ri = rsqrtf(fmaxf(di, 1.f));
        float ro = rsqrtf(fmaxf(dg, 1.f));
        dn_in[i] = ri; dn_out[i] = ro;
        float a0 = di * ro, a1 = dg * ro;         // h0 * dn_out
        x0[2 * i] = a0; x0[2 * i + 1] = a1;
        agg1[2 * i] = a0; agg1[2 * i + 1] = a1;   // self-loop contribution
        atomicAdd(&lc[n2g[i]], 1.f);
    }
    __syncthreads();
    if (t < BGR) atomicAdd(&cnt[t], lc[t]);
}

__global__ void k_edge1(const int* __restrict__ src, const int* __restrict__ dst,
                        const float* __restrict__ x0, float* agg1, int e) {
    int i = blockIdx.x * blockDim.x + threadIdx.x;
    if (i < e) {
        int s = src[i], d = dst[i];
        atomicAdd(&agg1[2 * d],     x0[2 * s]);
        atomicAdd(&agg1[2 * d + 1], x0[2 * s + 1]);
    }
}

// h1 = elu((agg1*dn_in) @ W1[2x128] + b1); y1 = h1*dn_out; agg2 seeded with self-loop (y1)
__global__ void k_hidden1(const float* __restrict__ agg1, const float* __restrict__ dn_in,
                          const float* __restrict__ dn_out,
                          const float* __restrict__ W1, const float* __restrict__ b1,
                          float* y1, float* agg2, int n) {
    int gid = blockIdx.x * blockDim.x + threadIdx.x;
    int i = gid >> 7, c = gid & 127;
    if (i >= n) return;
    float ri = dn_in[i];
    float a0 = agg1[2 * i] * ri, a1 = agg1[2 * i + 1] * ri;
    float v = fmaf(a0, W1[c], fmaf(a1, W1[128 + c], b1[c]));
    v = elu1(v);
    float y = v * dn_out[i];
    y1[i * 128 + c] = y;
    agg2[i * 128 + c] = y;
}

// the bandwidth hog: one wave per edge, gather 128 floats from y1[src], atomic-add into agg2[dst].
// edge index is wave-uniform so src/dst resolve to scalar loads.
__global__ __launch_bounds__(256) void k_edge2(const int* __restrict__ src,
                                               const int* __restrict__ dst,
                                               const float* __restrict__ y1,
                                               float* agg2, int e) {
    int edge = blockIdx.x * 8 + (threadIdx.x >> 5);
    if (edge >= e) return;
    int lane = threadIdx.x & 31;
    int s = src[edge], d = dst[edge];
    const float4 v = *(const float4*)(y1 + s * 128 + lane * 4);
    float* o = agg2 + d * 128 + lane * 4;
    atomicAdd(o + 0, v.x);
    atomicAdd(o + 1, v.y);
    atomicAdd(o + 2, v.z);
    atomicAdd(o + 3, v.w);
}

// one-time: W2 (fp32 128x128) -> bf16 B-fragment image (8192 dwords) in workspace.
// frag f = (kt*8+nt)*32+lane; dword v holds K pair 16*half+2v, +1 (contiguous k order, same as A).
__global__ void k_swz(const float* __restrict__ W2, unsigned* __restrict__ wsB) {
    int t = blockIdx.x * blockDim.x + threadIdx.x;   // 8192 threads
    if (t >= 8192) return;
    int f = t >> 3, v = t & 7;
    int l = f & 31, nt = (f >> 5) & 7, kt = f >> 8;
    int half = l >> 4, nn = nt * 16 + (l & 15);
    int k = kt * 32 + half * 16 + 2 * v;
    unsigned lo = f2bfu(W2[k * 128 + nn]);
    unsigned hi = f2bfu(W2[(k + 1) * 128 + nn]);
    wsB[t] = lo | (hi << 16);
}

// WMMA GEMM: h2 = elu((agg2*dn_in) @ W2[128x128] + b2), fused per-graph sum pooling.
// 256 threads = 8 waves; wave handles 16 node rows x 128 output cols (8 col tiles, K=128 in 4 WMMAs).
__global__ __launch_bounds__(256) void k_gemm2(const float* __restrict__ agg2,
                                               const float* __restrict__ dn_in,
                                               const int* __restrict__ n2g,
                                               const unsigned* __restrict__ wsB,
                                               const float* __restrict__ b2,
                                               float* hg, int n) {
    __shared__ __align__(32) unsigned ldsB[4 * 8 * 32 * 8]; // pre-swizzled B fragments (32KB)
    __shared__ float hgl[BGR * 128];                        // per-block graph pool accumulator

    int tid = threadIdx.x;
    for (int i = tid; i < BGR * 128; i += 256) hgl[i] = 0.f;

#if __has_builtin(__builtin_amdgcn_tensor_load_to_lds)
    // Tensor Data Mover: DMA the 32KB fragment image global->LDS (1-D tile, data_size=4B).
    if (tid < 32) {
        unsigned lds_off = (unsigned)(unsigned long long)(&ldsB[0]);
        unsigned long long ga = (unsigned long long)wsB;
        uv4 g0;
        g0[0] = 1u;                                            // count=1 (valid descriptor)
        g0[1] = lds_off;                                       // lds_addr (bytes)
        g0[2] = (unsigned)ga;                                  // global_addr[31:0]
        g0[3] = (unsigned)((ga >> 32) & 0x1FFFFFFu) | (2u << 30); // global_addr[56:32] | type=2
        iv8 g1;
        g1[0] = 0x00020000;        // workgroup_mask=0, data_size=2 (4B)
        g1[1] = 0x20000000;        // tensor_dim0 low16 (8192) << 16
        g1[2] = 0x00010000;        // tensor_dim0 hi16=0 | tensor_dim1 low16=1
        g1[3] = 0x20000000;        // tensor_dim1 hi16=0 | tile_dim0=8192
        g1[4] = 0x00000001;        // tile_dim1=1, tile_dim2=0
        g1[5] = 8192;              // tensor_dim0_stride low
        g1[6] = 0;                 // tensor_dim0_stride hi | tensor_dim1_stride low
        g1[7] = 0;
        iv4 z; z[0] = 0; z[1] = 0; z[2] = 0; z[3] = 0;
#if __clang_major__ >= 23
        iv8 z8; for (int q = 0; q < 8; ++q) z8[q] = 0;
        __builtin_amdgcn_tensor_load_to_lds(g0, g1, z, z, z8, 0);
#else
        __builtin_amdgcn_tensor_load_to_lds(g0, g1, z, z, 0);
#endif
        __builtin_amdgcn_s_wait_tensorcnt(0);
    }
#else
    // fallback: plain vectorized LDS fill
    for (int i = tid * 4; i < 8192; i += 256 * 4)
        *(float4*)&ldsB[i] = *(const float4*)&wsB[i];
#endif
    __syncthreads();

    int wave = tid >> 5, lane = tid & 31;
    int half = lane >> 4, m = lane & 15;
    int R = blockIdx.x * 128 + wave * 16;
    int row = R + m;
    float dn = (row < n) ? dn_in[row] : 0.f;
    const float* rp = agg2 + row * 128;

    // A fragments (bf16 16x32 layout): built once, reused across all 8 column tiles
    v16bf afrag[4];
    #pragma unroll
    for (int kt = 0; kt < 4; ++kt) {
        v16bf a;
        #pragma unroll
        for (int v = 0; v < 8; ++v) {
            int k0 = (2 * v + half * 16) + kt * 32;
            float2 pr = make_float2(0.f, 0.f);
            if (row < n) pr = *(const float2*)(rp + k0);
            a[2 * v]     = f2bf(pr.x * dn);
            a[2 * v + 1] = f2bf(pr.y * dn);
        }
        afrag[kt] = a;
    }

    int ncol = lane & 15;
    for (int nt = 0; nt < 8; ++nt) {
        v8f c = {};
        #pragma unroll
        for (int kt = 0; kt < 4; ++kt) {
            const v16bf* bp = (const v16bf*)&ldsB[((kt * 8 + nt) * 32 + lane) * 8];
            c = __builtin_amdgcn_wmma_f32_16x16x32_bf16(false, afrag[kt], false, *bp,
                                                        (short)0, c, false, false);
        }
        int col = nt * 16 + ncol;
        float bias = b2[col];
        #pragma unroll
        for (int r = 0; r < 8; ++r) {
            int rr = R + half * 8 + r;       // C layout: lanes 0-15 -> M=r, lanes 16-31 -> M=8+r
            if (rr < n) {
                float v = elu1(c[r] + bias);
                atomicAdd(&hgl[n2g[rr] * 128 + col], v);
            }
        }
    }
    __syncthreads();
    for (int i = tid; i < BGR * 128; i += 256) atomicAdd(&hg[i], hgl[i]);
}

// ---------------- tiny head: embeds + 4-layer transformer + MLP, one workgroup ----------------

struct PP { const float* p[96]; };

__global__ __launch_bounds__(256) void k_head(PP pp, const int* __restrict__ algo_ops,
                                              const int* __restrict__ schedule,
                                              const float* __restrict__ hg,
                                              const float* __restrict__ cnt,
                                              float* out) {
    extern __shared__ float sm[];
    float* xb  = sm;            // 3072  x tokens [8][12][32]
    float* qb  = xb + 3072;     // 3072
    float* kb  = qb + 3072;     // 3072
    float* vb  = kb + 3072;     // 3072
    float* ob  = vb + 3072;     // 3072
    float* att = ob + 3072;     // 4608  [8][4][12][12]
    float* ffh = att + 4608;    // 3072  FF hidden per batch
    float* m2b = ffh + 3072;    // 512
    int tid = threadIdx.x;

    // ---- algo embed: a[8][96] into qb ----
    for (int i = tid; i < 8 * 96; i += 256) {
        int b = i / 96, d = i % 96;
        const float* tab; int idx;
        if (d < 32)      { tab = pp.p[5]; idx = algo_ops[b * 3 + 0]; }
        else if (d < 64) { tab = pp.p[6]; idx = algo_ops[b * 3 + 1]; d -= 32; }
        else             { tab = pp.p[7]; idx = algo_ops[b * 3 + 2]; d -= 64; }
        qb[i] = tab[idx * 32 + d];
    }
    __syncthreads();
    for (int i = tid; i < 8 * 128; i += 256) {          // relu(a @ algo1)
        int b = i >> 7, c = i & 127;
        float s = pp.p[9][c];
        const float* W = pp.p[8];
        for (int j = 0; j < 96; ++j) s += qb[b * 96 + j] * W[j * 128 + c];
        kb[i] = fmaxf(s, 0.f);
    }
    __syncthreads();
    for (int i = tid; i < 8 * 128; i += 256) {          // algo_feat -> tokens 0..3
        int b = i >> 7, c = i & 127;
        float s = pp.p[11][c];
        const float* W = pp.p[10];
        for (int j = 0; j < 128; ++j) s += kb[b * 128 + j] * W[j * 128 + c];
        xb[b * 384 + (c >> 5) * 32 + (c & 31)] = s;
    }
    // ---- graph tokens 4..7 ----
    for (int i = tid; i < 8 * 128; i += 256) {
        int b = i >> 7, c = i & 127;
        float cn = fmaxf(cnt[b], 1.f);
        float s = pp.p[17][c];
        const float* W = pp.p[16];
        for (int j = 0; j < 128; ++j) s += (hg[b * 128 + j] / cn) * W[j * 128 + c];
        xb[b * 384 + (4 + (c >> 5)) * 32 + (c & 31)] = s;
    }
    // ---- sched embed -> tokens 8..11 ----
    for (int i = tid; i < 8 * 128; i += 256) {
        int b = i >> 7, d = i & 127;
        const float* tab = pp.p[18 + (d >> 5)];
        int idx = schedule[b * 4 + (d >> 5)];
        vb[i] = tab[idx * 32 + (d & 31)];
    }
    __syncthreads();
    for (int i = tid; i < 8 * 128; i += 256) {
        int b = i >> 7, c = i & 127;
        float s = pp.p[23][c];
        const float* W = pp.p[22];
        for (int j = 0; j < 128; ++j) s += vb[b * 128 + j] * W[j * 128 + c];
        ob[i] = fmaxf(s, 0.f);
    }
    __syncthreads();
    for (int i = tid; i < 8 * 128; i += 256) {
        int b = i >> 7, c = i & 127;
        float s = pp.p[25][c];
        const float* W = pp.p[24];
        for (int j = 0; j < 128; ++j) s += ob[b * 128 + j] * W[j * 128 + c];
        xb[b * 384 + (8 + (c >> 5)) * 32 + (c & 31)] = s;
    }
    __syncthreads();

    // ---- 4 post-norm transformer layers ----
    for (int L = 0; L < 4; ++L) {
        const float* const* lp = &pp.p[26 + L * 16];
        // q,k,v
        for (int i = tid; i < 8 * 12 * 32; i += 256) {
            int bt = i >> 5, d = i & 31;
            const float* xr = &xb[bt * 32];
            float sq = lp[1][d], sk = lp[3][d], sv = lp[5][d];
            for (int j = 0; j < 32; ++j) {
                float x = xr[j];
                sq += x * lp[0][j * 32 + d];
                sk += x * lp[2][j * 32 + d];
                sv += x * lp[4][j * 32 + d];
            }
            qb[i] = sq; kb[i] = sk; vb[i] = sv;
        }
        __syncthreads();
        // scores + softmax; one row per (b,h,tq)
        for (int r = tid; r < 384; r += 256) {
            int b = r / 48, h = (r / 12) % 4, tq = r % 12;
            const float* q = &qb[(b * 12 + tq) * 32 + h * 8];
            float sc[12]; float mx = -1e30f;
            for (int tk = 0; tk < 12; ++tk) {
                const float* kk = &kb[(b * 12 + tk) * 32 + h * 8];
                float s = 0.f;
                for (int e = 0; e < 8; ++e) s += q[e] * kk[e];
                s *= 0.35355339059327373f;   // 1/sqrt(8)
                sc[tk] = s; mx = fmaxf(mx, s);
            }
            float den = 0.f;
            for (int tk = 0; tk < 12; ++tk) { sc[tk] = expf(sc[tk] - mx); den += sc[tk]; }
            float inv = 1.f / den;
            float* ar = &att[((b * 4 + h) * 12 + tq) * 12];
            for (int tk = 0; tk < 12; ++tk) ar[tk] = sc[tk] * inv;
        }
        __syncthreads();
        // o = att @ v
        for (int i = tid; i < 8 * 12 * 32; i += 256) {
            int bt = i >> 5, d = i & 31, b = bt / 12, tq = bt % 12, h = d >> 3;
            const float* ar = &att[((b * 4 + h) * 12 + tq) * 12];
            float s = 0.f;
            for (int tk = 0; tk < 12; ++tk) s += ar[tk] * vb[(b * 12 + tk) * 32 + d];
            ob[i] = s;
        }
        __syncthreads();
        // residual + output proj -> qb
        for (int i = tid; i < 8 * 12 * 32; i += 256) {
            int bt = i >> 5, d = i & 31;
            float s = lp[7][d];
            const float* orow = &ob[bt * 32];
            for (int j = 0; j < 32; ++j) s += orow[j] * lp[6][j * 32 + d];
            qb[i] = xb[i] + s;
        }
        __syncthreads();
        // ln1 -> xb
        for (int r = tid; r < 96; r += 256) {
            const float* y = &qb[r * 32];
            float mean = 0.f;
            for (int j = 0; j < 32; ++j) mean += y[j];
            mean *= (1.f / 32.f);
            float var = 0.f;
            for (int j = 0; j < 32; ++j) { float dd = y[j] - mean; var += dd * dd; }
            var *= (1.f / 32.f);
            float inv = rsqrtf(var + 1e-5f);
            for (int j = 0; j < 32; ++j)
                xb[r * 32 + j] = (y[j] - mean) * inv * lp[12][j] + lp[13][j];
        }
        __syncthreads();
        // feed-forward, per batch to bound LDS
        for (int b = 0; b < 8; ++b) {
            for (int i = tid; i < 12 * 256; i += 256) {
                int t = i >> 8, h = i & 255;
                const float* xr = &xb[(b * 12 + t) * 32];
                float s = lp[9][h];
                for (int j = 0; j < 32; ++j) s += xr[j] * lp[8][j * 256 + h];
                ffh[i] = fmaxf(s, 0.f);
            }
            __syncthreads();
            for (int i = tid; i < 12 * 32; i += 256) {
                int t = i >> 5, d = i & 31;
                float s = lp[11][d];
                for (int h = 0; h < 256; ++h) s += ffh[t * 256 + h] * lp[10][h * 32 + d];
                ob[(b * 12 + t) * 32 + d] = xb[(b * 12 + t) * 32 + d] + s;
            }
            __syncthreads();
        }
        // ln2 -> xb
        for (int r = tid; r < 96; r += 256) {
            const float* y = &ob[r * 32];
            float mean = 0.f;
            for (int j = 0; j < 32; ++j) mean += y[j];
            mean *= (1.f / 32.f);
            float var = 0.f;
            for (int j = 0; j < 32; ++j) { float dd = y[j] - mean; var += dd * dd; }
            var *= (1.f / 32.f);
            float inv = rsqrtf(var + 1e-5f);
            for (int j = 0; j < 32; ++j)
                xb[r * 32 + j] = (y[j] - mean) * inv * lp[14][j] + lp[15][j];
        }
        __syncthreads();
    }

    // ---- final MLP head ----
    for (int i = tid; i < 8 * 128; i += 256) {
        int b = i >> 7, c = i & 127;
        float s = pp.p[91][c];
        const float* W = pp.p[90];
        const float* xr = &xb[b * 384];
        for (int j = 0; j < 384; ++j) s += xr[j] * W[j * 128 + c];
        qb[i] = fmaxf(s, 0.f);
    }
    __syncthreads();
    for (int i = tid; i < 8 * 64; i += 256) {
        int b = i >> 6, c = i & 63;
        float s = pp.p[93][c];
        const float* W = pp.p[92];
        for (int j = 0; j < 128; ++j) s += qb[b * 128 + j] * W[j * 64 + c];
        m2b[i] = fmaxf(s, 0.f);
    }
    __syncthreads();
    for (int b = tid; b < 8; b += 256) {
        float s = pp.p[95][0];
        const float* W = pp.p[94];
        for (int j = 0; j < 64; ++j) s += m2b[b * 64 + j] * W[j];
        out[b] = s;
    }
}

// ---------------- launch ----------------

extern "C" void kernel_launch(void* const* d_in, const int* in_sizes, int n_in,
                              void* d_out, int out_size, void* d_ws, size_t ws_size,
                              hipStream_t stream) {
    const int* algo_ops = (const int*)d_in[0];
    const int* schedule = (const int*)d_in[1];
    const int* esrc     = (const int*)d_in[2];
    const int* edst     = (const int*)d_in[3];
    const int* n2g      = (const int*)d_in[4];
    const int E = in_sizes[2];
    const int N = in_sizes[4];

    float* ws = (float*)d_ws;
    float* in_deg  = ws; ws += N;
    float* out_deg = ws; ws += N;
    float* dn_in   = ws; ws += N;
    float* dn_out  = ws; ws += N;
    float* x0      = ws; ws += 2 * (size_t)N;
    float* agg1    = ws; ws += 2 * (size_t)N;
    float* y1      = ws; ws += 128 * (size_t)N;
    float* agg2    = ws; ws += 128 * (size_t)N;
    float* hg      = ws; ws += BGR * 128;
    float* cnt     = ws; ws += BGR;
    unsigned* wsB  = (unsigned*)ws; ws += 8192;   // swizzled bf16 W2 fragment image

    const float* gc1W = (const float*)d_in[12];
    const float* gc1b = (const float*)d_in[13];
    const float* gc2W = (const float*)d_in[14];
    const float* gc2b = (const float*)d_in[15];

    k_init<<<(N + 255) / 256, 256, 0, stream>>>(in_deg, out_deg, hg, cnt, N);
    k_deg<<<(E + 255) / 256, 256, 0, stream>>>(esrc, edst, in_deg, out_deg, E);
    k_nodeprep<<<(N + 255) / 256, 256, 0, stream>>>(in_deg, out_deg, n2g,
                                                    dn_in, dn_out, x0, agg1, cnt, N);
    k_edge1<<<(E + 255) / 256, 256, 0, stream>>>(esrc, edst, x0, agg1, E);
    k_hidden1<<<(N * 128 + 255) / 256, 256, 0, stream>>>(agg1, dn_in, dn_out,
                                                         gc1W, gc1b, y1, agg2, N);
    k_edge2<<<(E + 7) / 8, 256, 0, stream>>>(esrc, edst, y1, agg2, E);
    k_swz<<<32, 256, 0, stream>>>(gc2W, wsB);
    k_gemm2<<<(N + 127) / 128, 256, 0, stream>>>(agg2, dn_in, n2g, wsB, gc2b, hg, N);

    PP pp;
    for (int i = 0; i < 96; ++i) pp.p[i] = (i < n_in) ? (const float*)d_in[i] : nullptr;
    size_t smem = (3072 * 5 + 4608 + 3072 + 512) * sizeof(float);
    k_head<<<1, 256, smem, stream>>>(pp, algo_ops, schedule, hg, cnt, (float*)d_out);
}
